// Intra_Modal_Modeling_62852551410314
// MI455X (gfx1250) — matrix operand order repaired and checked by reading, not compiled, not run on hardware
//
#include <hip/hip_runtime.h>
#include <hip/hip_bf16.h>
#include <cstddef>
#include <cstdint>

typedef __bf16 bhalf;
typedef __attribute__((ext_vector_type(16))) __bf16 v16bf;
typedef __attribute__((ext_vector_type(8)))  __bf16 v8bf;
typedef __attribute__((ext_vector_type(8)))  float  v8f;
typedef int v4i_ __attribute__((vector_size(16)));

union FragU { v16bf v; v8bf h[2]; };

// ---- gfx1250 async global->LDS path (guarded; falls back to ds_store) ----
#if defined(__has_builtin)
#if __has_builtin(__builtin_amdgcn_global_load_async_to_lds_b128)
#define HAVE_ASYNC_LDS 1
#endif
#endif
#ifndef HAVE_ASYNC_LDS
#define HAVE_ASYNC_LDS 0
#endif

__device__ __forceinline__ void async_wait0() {
#if defined(__has_builtin)
#if __has_builtin(__builtin_amdgcn_s_wait_asynccnt)
  __builtin_amdgcn_s_wait_asynccnt(0);
#else
  asm volatile("s_wait_asynccnt 0x0" ::: "memory");
#endif
#else
  asm volatile("s_wait_asynccnt 0x0" ::: "memory");
#endif
}

#if HAVE_ASYNC_LDS
// 16-byte async copy: global (AS1) -> LDS (AS3), tracked by ASYNCcnt
__device__ __forceinline__ void async_copy16(const bhalf* gsrc, bhalf* ldst) {
  __builtin_amdgcn_global_load_async_to_lds_b128(
      (__attribute__((address_space(1))) v4i_*)gsrc,
      (__attribute__((address_space(3))) v4i_*)ldst, 0, 0);
}
#endif

__device__ __forceinline__ v8f wmma_bf(v16bf a, v16bf b, v8f c) {
  // D = A(16x32 bf16) * B(32x16 bf16) + C(16x16 f32)
  return __builtin_amdgcn_wmma_f32_16x16x32_bf16(false, a, false, b, (short)0, c, false, false);
}

__device__ __forceinline__ float gelu_f(float x) {
  return 0.5f * x * (1.0f + erff(x * 0.70710678118654752440f));
}
__device__ __forceinline__ float wsum32(float v) {
#pragma unroll
  for (int o = 1; o < 32; o <<= 1) v += __shfl_xor(v, o, 32);
  return v;
}
__device__ __forceinline__ float hsum16(float v) {
#pragma unroll
  for (int o = 1; o < 16; o <<= 1) v += __shfl_xor(v, o, 32);
  return v;
}
__device__ __forceinline__ float hmax16(float v) {
#pragma unroll
  for (int o = 1; o < 16; o <<= 1) v = fmaxf(v, __shfl_xor(v, o, 32));
  return v;
}

// ---------------------------------------------------------------------------
// fp32 -> bf16 conversion (weights / activations)
// ---------------------------------------------------------------------------
__global__ void cvt_kernel(const float* __restrict__ s, bhalf* __restrict__ d, size_t n) {
  size_t i = (size_t)blockIdx.x * blockDim.x + threadIdx.x;
  const size_t stride = (size_t)gridDim.x * blockDim.x;
  for (; i < n; i += stride) d[i] = (bhalf)s[i];
}

// h = x + pos (broadcast over batch) -> bf16 ; also bf16(x)
__global__ void addpos_kernel(const float* __restrict__ X, const float* __restrict__ P,
                              bhalf* __restrict__ Hb, bhalf* __restrict__ Xb,
                              size_t nd, size_t total) {
  size_t i = (size_t)blockIdx.x * blockDim.x + threadIdx.x;
  const size_t stride = (size_t)gridDim.x * blockDim.x;
  for (; i < total; i += stride) {
    float x = X[i];
    Xb[i] = (bhalf)x;
    Hb[i] = (bhalf)(x + P[i % nd]);
  }
}

// out = x + y (fp32, optional bf16 copy)
__global__ void addres_kernel(const float* __restrict__ X, const float* __restrict__ Y,
                              float* __restrict__ Of, bhalf* __restrict__ Ob, size_t total) {
  size_t i = (size_t)blockIdx.x * blockDim.x + threadIdx.x;
  const size_t stride = (size_t)gridDim.x * blockDim.x;
  for (; i < total; i += stride) {
    float v = X[i] + Y[i];
    Of[i] = v;
    if (Ob) Ob[i] = (bhalf)v;
  }
}

// ---------------------------------------------------------------------------
// WMMA GEMM:  C[M,Nout] = A[M,K](bf16) * W[Nout,K]^T(bf16) + bias
// block = 128 thr (4 waves); block tile 128x64, wave tile 32x64, K-step 32.
// Weight tile staged in LDS (async global->LDS when available); each wave
// runs 8 back-to-back WMMAs per K-step against the 4 shared B-fragments.
// Epilogue: bias, optional GELU, alpha, fp32 store, bf16 store (optionally
// transposed per-head [b][h][hd][tok] for the attention P*V operand).
// ---------------------------------------------------------------------------
__global__ __launch_bounds__(128)
void gemm_bias_kernel(const bhalf* __restrict__ A, const bhalf* __restrict__ W,
                      const float* __restrict__ bias,
                      float* __restrict__ Cf, bhalf* __restrict__ Cb,
                      int K, int Nout, float alpha, int do_gelu,
                      int trans_nh, int Ntok) {
  __shared__ bhalf Bt[64 * 32];
  const int lane = threadIdx.x & 31;
  const int wave = threadIdx.x >> 5;
  const int grp  = lane >> 4;
  const int l16  = lane & 15;
  const int n0 = blockIdx.x * 64;
  const int m0 = blockIdx.y * 128 + wave * 32;

  v8f acc[2][4];
#pragma unroll
  for (int hfi = 0; hfi < 2; hfi++)
#pragma unroll
    for (int t = 0; t < 4; t++)
#pragma unroll
      for (int i = 0; i < 8; i++) acc[hfi][t][i] = 0.0f;

  const bhalf* Abase0 = A + (size_t)(m0 + l16) * K;
  const bhalf* Abase1 = A + (size_t)(m0 + 16 + l16) * K;

  for (int k0 = 0; k0 < K; k0 += 32) {
    {
      const int row  = threadIdx.x >> 1;   // 0..63 (weight row within tile)
      const int half = threadIdx.x & 1;    // which 16-element K chunk
      const bhalf* src = W + (size_t)(n0 + row) * K + k0 + half * 16;
      bhalf* dst = &Bt[row * 32 + half * 16];
#if HAVE_ASYNC_LDS
      async_copy16(src, dst);
      async_copy16(src + 8, dst + 8);
      async_wait0();
#else
      v8bf w0 = *(const v8bf*)(src);
      v8bf w1 = *(const v8bf*)(src + 8);
      *(v8bf*)(dst)     = w0;
      *(v8bf*)(dst + 8) = w1;
#endif
    }
    __syncthreads();
    // A-frags: lanes 0-15 K {0-7,16-23}; lanes 16-31 K {8-15,24-31}
    FragU af0, af1;
    {
      const bhalf* ap0 = Abase0 + k0 + grp * 8;
      const bhalf* ap1 = Abase1 + k0 + grp * 8;
      af0.h[0] = *(const v8bf*)(ap0);
      af0.h[1] = *(const v8bf*)(ap0 + 16);
      af1.h[0] = *(const v8bf*)(ap1);
      af1.h[1] = *(const v8bf*)(ap1 + 16);
    }
    if (k0 + 32 < K) {
      __builtin_prefetch(Abase0 + k0 + 32, 0, 3);
      __builtin_prefetch(Abase1 + k0 + 32, 0, 3);
    }
    // B-frags: lanes 0-15 hold K {0-7,16-23} (col = lane), lanes 16-31 K {8-15,24-31}
    FragU bfr[4];
#pragma unroll
    for (int t = 0; t < 4; t++) {
      const bhalf* bp = &Bt[(t * 16 + l16) * 32 + grp * 8];
      bfr[t].h[0] = *(const v8bf*)(bp);
      bfr[t].h[1] = *(const v8bf*)(bp + 16);
    }
#pragma unroll
    for (int t = 0; t < 4; t++) acc[0][t] = wmma_bf(af0.v, bfr[t].v, acc[0][t]);
#pragma unroll
    for (int t = 0; t < 4; t++) acc[1][t] = wmma_bf(af1.v, bfr[t].v, acc[1][t]);
    __syncthreads();
  }
#pragma unroll
  for (int hfi = 0; hfi < 2; hfi++) {
#pragma unroll
    for (int t = 0; t < 4; t++) {
      const int col = n0 + t * 16 + l16;
      const float bv = bias ? bias[col] : 0.0f;
#pragma unroll
      for (int r = 0; r < 8; r++) {
        const int row = m0 + hfi * 16 + grp * 8 + r;
        float v = acc[hfi][t][r] + bv;
        if (do_gelu) v = gelu_f(v);
        v *= alpha;
        if (Cf) Cf[(size_t)row * Nout + col] = v;
        if (Cb) {
          if (trans_nh > 0) {
            const int HD = Nout / trans_nh;
            const int b = row / Ntok, tok = row - b * Ntok;
            const int h = col / HD, d = col - h * HD;
            Cb[(((size_t)b * trans_nh + h) * HD + d) * Ntok + tok] = (bhalf)v;
          } else {
            Cb[(size_t)row * Nout + col] = (bhalf)v;
          }
        }
      }
    }
  }
}

// ---------------------------------------------------------------------------
// Flash attention. grid = (N/64, nh, B), 4 waves, 16 q-rows per wave.
// MODE 0: key-padding bias from mask[B,N]
// MODE 1: keep key j iff sim_sim(b,q,j) >= thr[b,q]   (top-k largest)
// MODE 2: keep key j iff sim_unsim(b,q,j) <= thr[b,q] (top-k smallest)
// ---------------------------------------------------------------------------
template <int HD, int MODE>
__global__ __launch_bounds__(128)
void attn_kernel(const bhalf* __restrict__ Q, const bhalf* __restrict__ Kb,
                 const bhalf* __restrict__ Vt, bhalf* __restrict__ Ob,
                 int N, int D, int nh,
                 const unsigned char* __restrict__ mask,
                 const float* __restrict__ sim,
                 const unsigned char* __restrict__ mm,
                 const float* __restrict__ thr) {
  constexpr int NKC = HD / 32;  // K-chunks for Q*K^T
  constexpr int NOT = HD / 16;  // output tiles
  __shared__ bhalf Pl[4][16 * 32];
  const int lane = threadIdx.x & 31;
  const int wave = threadIdx.x >> 5;
  const int grp = lane >> 4, l16 = lane & 15;
  const int b = blockIdx.z, h = blockIdx.y;
  const int q0 = blockIdx.x * 64 + wave * 16;
  const int hoff = h * HD;
  bhalf* Pw = &Pl[wave][0];

  FragU qf[NKC];
  {
    const bhalf* qp = Q + (size_t)(b * N + q0 + l16) * D + hoff + grp * 8;
#pragma unroll
    for (int c = 0; c < NKC; c++) {
      qf[c].h[0] = *(const v8bf*)(qp + c * 32);
      qf[c].h[1] = *(const v8bf*)(qp + c * 32 + 16);
    }
  }
  v8f oacc[NOT];
#pragma unroll
  for (int t = 0; t < NOT; t++)
#pragma unroll
    for (int i = 0; i < 8; i++) oacc[t][i] = 0.0f;
  float mi[8], li[8], thr8[8];
#pragma unroll
  for (int r = 0; r < 8; r++) { mi[r] = -3.0e38f; li[r] = 0.0f; thr8[r] = 0.0f; }
  if (MODE != 0) {
#pragma unroll
    for (int r = 0; r < 8; r++) thr8[r] = thr[(size_t)b * N + q0 + grp * 8 + r];
  }

  for (int kb = 0; kb < N; kb += 32) {
    v8f s[2];
#pragma unroll
    for (int t = 0; t < 2; t++) {
#pragma unroll
      for (int i = 0; i < 8; i++) s[t][i] = 0.0f;
      const bhalf* kp = Kb + (size_t)(b * N + kb + t * 16 + l16) * D + hoff + grp * 8;
#pragma unroll
      for (int c = 0; c < NKC; c++) {
        FragU kf;
        kf.h[0] = *(const v8bf*)(kp + c * 32);
        kf.h[1] = *(const v8bf*)(kp + c * 32 + 16);
        s[t] = wmma_bf(qf[c].v, kf.v, s[t]);
      }
    }
    // bias
#pragma unroll
    for (int t = 0; t < 2; t++) {
      const int kj = kb + t * 16 + l16;
      if (MODE == 0) {
        const float bv = mask[(size_t)b * N + kj] ? -1e9f : 0.0f;
#pragma unroll
        for (int r = 0; r < 8; r++) s[t][r] += bv;
      } else {
#pragma unroll
        for (int r = 0; r < 8; r++) {
          const int row = q0 + grp * 8 + r;
          const size_t ix = ((size_t)b * N + row) * N + kj;
          const float sv = sim[ix];
          const bool mv = mm[ix] != 0;
          float bv;
          if (MODE == 1) {
            float ss = (mv ? sv : -1.0f) - (row == kj ? 1.0f : 0.0f);
            bv = (ss >= thr8[r]) ? 0.0f : -1e9f;
          } else {
            float su = mv ? sv : 2.0f;
            bv = (su <= thr8[r]) ? 0.0f : -1e9f;
          }
          s[t][r] += bv;
        }
      }
    }
    // online softmax over this 32-key block
    float psc[8];
#pragma unroll
    for (int r = 0; r < 8; r++) {
      float v = fmaxf(s[0][r], s[1][r]);
      v = hmax16(v);
      float mnew = fmaxf(mi[r], v);
      psc[r] = __expf(mi[r] - mnew);
      mi[r] = mnew;
    }
#pragma unroll
    for (int t = 0; t < NOT; t++)
#pragma unroll
      for (int r = 0; r < 8; r++) oacc[t][r] *= psc[r];
#pragma unroll
    for (int r = 0; r < 8; r++) {
      float p0 = __expf(s[0][r] - mi[r]);
      float p1 = __expf(s[1][r] - mi[r]);
      s[0][r] = p0; s[1][r] = p1;
      li[r] = li[r] * psc[r] + hsum16(p0 + p1);
    }
    // repack P from C-layout to A-frag layout via wave-private LDS
#pragma unroll
    for (int t = 0; t < 2; t++)
#pragma unroll
      for (int r = 0; r < 8; r++)
        Pw[(grp * 8 + r) * 32 + t * 16 + l16] = (bhalf)s[t][r];
    FragU pf;
    {
      const bhalf* pp = Pw + l16 * 32 + grp * 8;
      pf.h[0] = *(const v8bf*)(pp);
      pf.h[1] = *(const v8bf*)(pp + 16);
    }
    // O += P * V   (V stored transposed [b][h][hd][tok] -> contiguous B-frag)
#pragma unroll
    for (int t = 0; t < NOT; t++) {
      FragU vf;
      const bhalf* vp = Vt + ((size_t)(b * nh + h) * HD + t * 16 + l16) * N + kb + grp * 8;
      vf.h[0] = *(const v8bf*)(vp);
      vf.h[1] = *(const v8bf*)(vp + 16);
      oacc[t] = wmma_bf(pf.v, vf.v, oacc[t]);
    }
  }
#pragma unroll
  for (int t = 0; t < NOT; t++) {
    const int col = hoff + t * 16 + l16;
#pragma unroll
    for (int r = 0; r < 8; r++) {
      const int row = q0 + grp * 8 + r;
      Ob[(size_t)(b * N + row) * D + col] = (bhalf)(oacc[t][r] / li[r]);
    }
  }
}

// ---------------------------------------------------------------------------
// sim[b,i,j] = clip((nv_i . nv_j + 1) * 0.5, 0, 1)  -- batched WMMA, K=64
// ---------------------------------------------------------------------------
__global__ __launch_bounds__(128)
void sim_kernel(const bhalf* __restrict__ NV, float* __restrict__ S, int N) {
  const int lane = threadIdx.x & 31, wave = threadIdx.x >> 5;
  const int grp = lane >> 4, l16 = lane & 15;
  const int b = blockIdx.z;
  const int i0 = blockIdx.y * 64 + wave * 16;
  const int j0 = blockIdx.x * 64;
  const bhalf* Ab = NV + (size_t)b * N * 64;
  v8f acc[4];
#pragma unroll
  for (int t = 0; t < 4; t++)
#pragma unroll
    for (int i = 0; i < 8; i++) acc[t][i] = 0.0f;
#pragma unroll
  for (int k0 = 0; k0 < 64; k0 += 32) {
    FragU af;
    const bhalf* ap = Ab + (size_t)(i0 + l16) * 64 + k0 + grp * 8;
    af.h[0] = *(const v8bf*)(ap);
    af.h[1] = *(const v8bf*)(ap + 16);
#pragma unroll
    for (int t = 0; t < 4; t++) {
      FragU bf;
      const bhalf* bp = Ab + (size_t)(j0 + t * 16 + l16) * 64 + k0 + grp * 8;
      bf.h[0] = *(const v8bf*)(bp);
      bf.h[1] = *(const v8bf*)(bp + 16);
      acc[t] = wmma_bf(af.v, bf.v, acc[t]);
    }
  }
#pragma unroll
  for (int t = 0; t < 4; t++) {
    const int col = j0 + t * 16 + l16;
#pragma unroll
    for (int r = 0; r < 8; r++) {
      const int row = i0 + grp * 8 + r;
      float v = (acc[t][r] + 1.0f) * 0.5f;
      v = fminf(fmaxf(v, 0.0f), 1.0f);
      S[((size_t)b * N + row) * N + col] = v;
    }
  }
}

// ---------------------------------------------------------------------------
// LayerNorm (optional residual, optional GELU), one wave per row.
// ---------------------------------------------------------------------------
template <int F>
__global__ __launch_bounds__(32)
void ln_kernel(const float* __restrict__ X, const float* __restrict__ R,
               const float* __restrict__ g, const float* __restrict__ bt,
               float* __restrict__ Yf, bhalf* __restrict__ Yb, int do_gelu) {
  constexpr int C = F / 32;
  const size_t row = blockIdx.x;
  const int lane = threadIdx.x;
  const float* x = X + row * F;
  float vals[C];
  float s = 0.0f;
#pragma unroll
  for (int i = 0; i < C; i++) {
    float v = x[lane + i * 32];
    if (R) v += R[row * F + lane + i * 32];
    vals[i] = v;
    s += v;
  }
  s = wsum32(s);
  const float mean = s * (1.0f / F);
  float q = 0.0f;
#pragma unroll
  for (int i = 0; i < C; i++) {
    float d = vals[i] - mean;
    q += d * d;
  }
  q = wsum32(q);
  const float inv = rsqrtf(q * (1.0f / F) + 1e-5f);
#pragma unroll
  for (int i = 0; i < C; i++) {
    const int col = lane + i * 32;
    float y = (vals[i] - mean) * inv * g[col] + bt[col];
    if (do_gelu) y = gelu_f(y);
    if (Yf) Yf[row * F + col] = y;
    if (Yb) Yb[row * F + col] = (bhalf)y;
  }
}

// normalize 64-wide rows of c -> bf16
__global__ __launch_bounds__(32)
void rownorm_kernel(const float* __restrict__ C, bhalf* __restrict__ NVb) {
  const size_t row = blockIdx.x;
  const int lane = threadIdx.x;
  const float v0 = C[row * 64 + lane];
  const float v1 = C[row * 64 + 32 + lane];
  float s = wsum32(v0 * v0 + v1 * v1);
  const float inv = rsqrtf(s);
  NVb[row * 64 + lane]      = (bhalf)(v0 * inv);
  NVb[row * 64 + 32 + lane] = (bhalf)(v1 * inv);
}

// fixed_k per batch = clip(min(64, count(~mask)-1), 1)  (float reduction)
__global__ __launch_bounds__(32)
void maskcount_kernel(const unsigned char* __restrict__ mask, int* __restrict__ kk, int N) {
  const int b = blockIdx.x, lane = threadIdx.x;
  float c = 0.0f;
  for (int j = lane; j < N; j += 32) c += (mask[(size_t)b * N + j] == 0) ? 1.0f : 0.0f;
  c = wsum32(c);
  if (lane == 0) {
    int valid = (int)(c + 0.5f) - 1;
    if (valid < 1) valid = 1;
    int fk = valid < 64 ? valid : 64;
    if (fk < 1) fk = 1;
    kk[b] = fk;
  }
}

// order-preserving float<->uint maps
__device__ __forceinline__ unsigned int ford(float f) {
  unsigned int u = __float_as_uint(f);
  return (u & 0x80000000u) ? ~u : (u | 0x80000000u);
}
__device__ __forceinline__ float funord(unsigned int u) {
  unsigned int b = (u & 0x80000000u) ? (u & 0x7fffffffu) : ~u;
  return __uint_as_float(b);
}

// bitwise radix select of the k-th largest value; each lane holds 32 values
// in registers; cross-lane counting uses the float reduction path only.
__device__ __forceinline__ unsigned int radix_kth32(const unsigned int (&va)[32], int k) {
  unsigned int prefix = 0, hm = 0;
  for (int bit = 31; bit >= 0; --bit) {
    const unsigned int bm = 1u << bit;
    float c = 0.0f;
#pragma unroll
    for (int i = 0; i < 32; i++)
      c += (((va[i] & hm) == prefix) && ((va[i] & bm) != 0u)) ? 1.0f : 0.0f;
    c = wsum32(c);
    const int ci = (int)(c + 0.5f);
    if (ci >= k) prefix |= bm; else k -= ci;
    hm |= bm;
  }
  return prefix;
}

// one wave per (b,q) row; all row data register-resident (N == 1024)
__global__ __launch_bounds__(32)
void topk_kernel(const float* __restrict__ sim, const unsigned char* __restrict__ mm,
                 const int* __restrict__ kk, float* __restrict__ thrS,
                 float* __restrict__ thrU, int N) {
  const int q = blockIdx.x, b = blockIdx.y, lane = threadIdx.x;
  const float* srow = sim + ((size_t)b * N + q) * N;
  const unsigned char* mrow = mm + ((size_t)b * N + q) * N;
  unsigned int us[32], uu[32];
#pragma unroll
  for (int i = 0; i < 32; i++) {
    const int j = lane + i * 32;
    const float sv = srow[j];
    const bool mv = mrow[j] != 0;
    const float ss = (mv ? sv : -1.0f) - (j == q ? 1.0f : 0.0f);
    const float su = mv ? sv : 2.0f;
    us[i] = ford(ss);
    uu[i] = ~ford(su);  // inverted order -> k-th smallest via k-th largest
  }
  const int k = kk[b];
  const unsigned int ts = radix_kth32(us, k);
  const unsigned int tu = radix_kth32(uu, k);
  if (lane == 0) {
    thrS[(size_t)b * N + q] = funord(ts);
    thrU[(size_t)b * N + q] = funord(~tu);
  }
}

// ---------------------------------------------------------------------------
// Host orchestration
// ---------------------------------------------------------------------------
extern "C" void kernel_launch(void* const* d_in, const int* in_sizes, int n_in,
                              void* d_out, int out_size, void* d_ws, size_t ws_size,
                              hipStream_t stream) {
  (void)in_sizes; (void)n_in; (void)out_size; (void)ws_size;
  const int Bb = 8, Nn = 1024, Dd = 512;
  const size_t BN = (size_t)Bb * Nn;          // 8192
  const size_t BND = BN * Dd;                 // 4194304
  const size_t ND = (size_t)Nn * Dd;          // 524288
  const size_t XE = BND;
  const size_t SIME = (size_t)Bb * Nn * Nn;   // 8388608

  float* out_x = (float*)d_out;
  float* out_sim = out_x + XE;
  float* out_c = out_sim + SIME;

  char* wsb = (char*)d_ws;
  size_t off = 0;
  auto alloc = [&](size_t bytes) -> void* {
    void* p = wsb + off;
    off += (bytes + 255) & ~(size_t)255;
    return p;
  };

  float* xf   = (float*)alloc(BND * 4);
  bhalf* xbf  = (bhalf*)alloc(BND * 2);
  bhalf* hbf  = (bhalf*)alloc(BND * 2);
  bhalf* qbf  = (bhalf*)alloc(BND * 2);
  bhalf* kbf  = (bhalf*)alloc(BND * 2);
  bhalf* vtbf = (bhalf*)alloc(BND * 2);
  bhalf* abf  = (bhalf*)alloc(BND * 2);
  bhalf* obf  = (bhalf*)alloc(BND * 2);
  float* f32a = (float*)alloc(BN * 1024 * 4);
  float* f32b = (float*)alloc(BND * 4);
  bhalf* bfa  = (bhalf*)alloc(BN * 1024 * 2);
  bhalf* nvbf = (bhalf*)alloc(BN * 64 * 2);
  float* thrS = (float*)alloc(BN * 4);
  float* thrU = (float*)alloc(BN * 4);
  int*   kkd  = (int*)alloc(Bb * 4);

  auto cvtw = [&](int idx, size_t n) -> bhalf* {
    bhalf* dst = (bhalf*)alloc(n * 2);
    int blocks = (int)((n + 255) / 256);
    if (blocks > 4096) blocks = 4096;
    cvt_kernel<<<blocks, 256, 0, stream>>>((const float*)d_in[idx], dst, n);
    return dst;
  };

  // ---- convert weights to bf16 ----
  bhalf *si_wq[3], *si_wk[3], *si_wv[3], *si_wo[3], *si_f1[3], *si_f2[3];
  for (int L = 0; L < 3; ++L) {
    const int base = 4 + 16 * L;
    si_wq[L] = cvtw(base + 0,  (size_t)Dd * Dd);
    si_wk[L] = cvtw(base + 2,  (size_t)Dd * Dd);
    si_wv[L] = cvtw(base + 4,  (size_t)Dd * Dd);
    si_wo[L] = cvtw(base + 6,  (size_t)Dd * Dd);
    si_f1[L] = cvtw(base + 10, (size_t)2 * Dd * Dd);
    si_f2[L] = cvtw(base + 12, (size_t)2 * Dd * Dd);
  }
  bhalf* ce_w1 = cvtw(52, 256 * 512);
  bhalf* ce_w2 = cvtw(56, 128 * 256);
  bhalf* ce_w3 = cvtw(60, 64 * 128);
  bhalf *ag_wq[2], *ag_wk[2], *ag_wv[2], *ag_wo[2], *ml_w1[2], *ml_w2[2], *ml_w3[2];
  for (int g = 0; g < 2; ++g) {
    const int ab = 62 + 18 * g, mb = 70 + 18 * g;
    ag_wq[g] = cvtw(ab + 0, (size_t)Dd * Dd);
    ag_wk[g] = cvtw(ab + 2, (size_t)Dd * Dd);
    ag_wv[g] = cvtw(ab + 4, (size_t)Dd * Dd);
    ag_wo[g] = cvtw(ab + 6, (size_t)Dd * Dd);
    ml_w1[g] = cvtw(mb + 0, (size_t)1024 * 512);
    ml_w2[g] = cvtw(mb + 4, (size_t)1024 * 1024);
    ml_w3[g] = cvtw(mb + 8, (size_t)512 * 1024);
  }

  auto gemm = [&](const bhalf* A, const bhalf* W, const float* bias,
                  float* Cf, bhalf* Cb, int K, int Nout,
                  float alpha, int gelu, int trans_nh) {
    dim3 grid(Nout / 64, (unsigned)(BN / 128));
    gemm_bias_kernel<<<grid, 128, 0, stream>>>(A, W, bias, Cf, Cb, K, Nout,
                                               alpha, gelu, trans_nh, Nn);
  };

  const unsigned char* maskp = (const unsigned char*)d_in[1];
  const float* posp = (const float*)d_in[2];
  const unsigned char* mmp = (const unsigned char*)d_in[3];
  const int ew_blocks = (int)((BND + 255) / 256);

  (void)hipMemcpyAsync(xf, d_in[0], BND * sizeof(float), hipMemcpyDeviceToDevice, stream);

  // ---- 3 Self_Interaction transformer blocks ----
  for (int L = 0; L < 3; ++L) {
    const int base = 4 + 16 * L;
    const float* bq  = (const float*)d_in[base + 1];
    const float* bk  = (const float*)d_in[base + 3];
    const float* bv  = (const float*)d_in[base + 5];
    const float* bo  = (const float*)d_in[base + 7];
    const float* l1g = (const float*)d_in[base + 8];
    const float* l1b = (const float*)d_in[base + 9];
    const float* fb1 = (const float*)d_in[base + 11];
    const float* fb2 = (const float*)d_in[base + 13];
    const float* l2g = (const float*)d_in[base + 14];
    const float* l2b = (const float*)d_in[base + 15];

    addpos_kernel<<<ew_blocks, 256, 0, stream>>>(xf, posp, hbf, xbf, ND, BND);
    gemm(hbf, si_wq[L], bq, nullptr, qbf, 512, 512, 0.125f, 0, 0);  // 1/sqrt(64)
    gemm(hbf, si_wk[L], bk, nullptr, kbf, 512, 512, 1.0f, 0, 0);
    gemm(xbf, si_wv[L], bv, nullptr, vtbf, 512, 512, 1.0f, 0, 8);   // [b][h][hd][tok]
    attn_kernel<64, 0><<<dim3(Nn / 64, 8, Bb), 128, 0, stream>>>(
        qbf, kbf, vtbf, abf, Nn, Dd, 8, maskp, nullptr, nullptr, nullptr);
    gemm(abf, si_wo[L], bo, f32b, nullptr, 512, 512, 1.0f, 0, 0);
    ln_kernel<512><<<(unsigned)BN, 32, 0, stream>>>(xf, f32b, l1g, l1b, xf, xbf, 0);
    gemm(xbf, si_f1[L], fb1, nullptr, bfa, 512, 1024, 1.0f, 1, 0);  // GELU
    gemm(bfa, si_f2[L], fb2, f32b, nullptr, 1024, 512, 1.0f, 0, 0);
    ln_kernel<512><<<(unsigned)BN, 32, 0, stream>>>(xf, f32b, l2g, l2b, xf, xbf, 0);
  }

  // ---- consist encoder: 512 -> 256 -> 128 -> 64 ----
  gemm(xbf, ce_w1, (const float*)d_in[53], f32b, nullptr, 512, 256, 1.0f, 0, 0);
  ln_kernel<256><<<(unsigned)BN, 32, 0, stream>>>(
      f32b, nullptr, (const float*)d_in[54], (const float*)d_in[55], nullptr, bfa, 1);
  gemm(bfa, ce_w2, (const float*)d_in[57], f32b, nullptr, 256, 128, 1.0f, 0, 0);
  ln_kernel<128><<<(unsigned)BN, 32, 0, stream>>>(
      f32b, nullptr, (const float*)d_in[58], (const float*)d_in[59], nullptr, bfa, 1);
  gemm(bfa, ce_w3, (const float*)d_in[61], out_c, nullptr, 128, 64, 1.0f, 0, 0);

  // ---- cosine similarity + top-k thresholds ----
  rownorm_kernel<<<(unsigned)BN, 32, 0, stream>>>(out_c, nvbf);
  sim_kernel<<<dim3(Nn / 64, Nn / 64, Bb), 128, 0, stream>>>(nvbf, out_sim, Nn);
  maskcount_kernel<<<Bb, 32, 0, stream>>>(maskp, kkd, Nn);
  topk_kernel<<<dim3(Nn, Bb), 32, 0, stream>>>(out_sim, mmp, kkd, thrS, thrU, Nn);

  // ---- 2 masked-MHA + MLP3 residual blocks ----
  for (int g = 0; g < 2; ++g) {
    const int ab = 62 + 18 * g, mb = 70 + 18 * g;
    const float* bq = (const float*)d_in[ab + 1];
    const float* bk = (const float*)d_in[ab + 3];
    const float* bv = (const float*)d_in[ab + 5];
    const float* bo = (const float*)d_in[ab + 7];
    const float* m_b1 = (const float*)d_in[mb + 1];
    const float* m_g1 = (const float*)d_in[mb + 2];
    const float* m_gb1 = (const float*)d_in[mb + 3];
    const float* m_b2 = (const float*)d_in[mb + 5];
    const float* m_g2 = (const float*)d_in[mb + 6];
    const float* m_gb2 = (const float*)d_in[mb + 7];
    const float* m_b3 = (const float*)d_in[mb + 9];

    gemm(xbf, ag_wq[g], bq, nullptr, qbf, 512, 512, 0.08838834764831845f, 0, 0);  // 1/sqrt(128)
    gemm(xbf, ag_wk[g], bk, nullptr, kbf, 512, 512, 1.0f, 0, 0);
    gemm(xbf, ag_wv[g], bv, nullptr, vtbf, 512, 512, 1.0f, 0, 4);
    if (g == 0)
      attn_kernel<128, 1><<<dim3(Nn / 64, 4, Bb), 128, 0, stream>>>(
          qbf, kbf, vtbf, abf, Nn, Dd, 4, nullptr, out_sim, mmp, thrS);
    else
      attn_kernel<128, 2><<<dim3(Nn / 64, 4, Bb), 128, 0, stream>>>(
          qbf, kbf, vtbf, abf, Nn, Dd, 4, nullptr, out_sim, mmp, thrU);
    gemm(abf, ag_wo[g], bo, nullptr, obf, 512, 512, 1.0f, 0, 0);
    gemm(obf, ml_w1[g], m_b1, f32a, nullptr, 512, 1024, 1.0f, 0, 0);
    ln_kernel<1024><<<(unsigned)BN, 32, 0, stream>>>(f32a, nullptr, m_g1, m_gb1, nullptr, bfa, 1);
    gemm(bfa, ml_w2[g], m_b2, f32a, nullptr, 1024, 1024, 1.0f, 0, 0);
    ln_kernel<1024><<<(unsigned)BN, 32, 0, stream>>>(f32a, nullptr, m_g2, m_gb2, nullptr, bfa, 1);
    gemm(bfa, ml_w3[g], m_b3, f32b, nullptr, 1024, 512, 1.0f, 0, 0);
    if (g == 0)
      addres_kernel<<<ew_blocks, 256, 0, stream>>>(xf, f32b, xf, xbf, BND);
    else
      addres_kernel<<<ew_blocks, 256, 0, stream>>>(xf, f32b, out_x, nullptr, BND);
  }
}
